// InstructionDecoder_6227702579419
// MI455X (gfx1250) — compile-verified, matrix operand order checked by reading, not run on hardware
//
#include <hip/hip_runtime.h>

typedef __attribute__((ext_vector_type(16))) __bf16 v16bf;
typedef __attribute__((ext_vector_type(8)))  __bf16 v8bf;
typedef __attribute__((ext_vector_type(4)))  __bf16 v4bf;
typedef __attribute__((ext_vector_type(8)))  float  v8f;
typedef __attribute__((ext_vector_type(4)))  float  v4f;
typedef __attribute__((ext_vector_type(4)))  int    v4i;

#define RS 40  // LDS row stride in bf16 elems: 32 data + 8 pad = 80B (16B aligned)

#if defined(__AMDGCN__) && __has_builtin(__builtin_amdgcn_global_load_async_to_lds_b128)
#define HAS_ASYNC_LDS 1
// pointer-to-int4 in global (AS1) / LDS (AS3) address spaces, per builtin signature
typedef __attribute__((address_space(1))) v4i* gv4i_p;
typedef __attribute__((address_space(3))) v4i* lv4i_p;
#else
#define HAS_ASYNC_LDS 0
#endif

// ---------------------------------------------------------------------------
// fp32 -> bf16 bulk convert (weights + x, done once up front)
// ---------------------------------------------------------------------------
__global__ void cvt_f32_bf16(const float* __restrict__ src, __bf16* __restrict__ dst, int n) {
  int i = (blockIdx.x * blockDim.x + threadIdx.x) * 4;
  if (i + 3 < n) {
    v4f f = *(const v4f*)(src + i);
    v4bf h;
    h[0] = (__bf16)f[0]; h[1] = (__bf16)f[1];
    h[2] = (__bf16)f[2]; h[3] = (__bf16)f[3];
    *(v4bf*)(dst + i) = h;
  }
}

// ---------------------------------------------------------------------------
// Stage one 128x32 bf16 tile into LDS.
// 256 threads: thread t -> row = t>>1, 16 cols from (t&1)*16  => 2x b128/thread.
// Async path: GLOBAL_LOAD_ASYNC_TO_LDS_B128 (ASYNCcnt), no VGPR round-trip.
// ---------------------------------------------------------------------------
__device__ __forceinline__ void stage_tile(const __bf16* __restrict__ g, int ld,
                                           int row0, int k0, __bf16* s, int tid) {
  const int r  = tid >> 1;
  const int c0 = (tid & 1) << 4;
  const __bf16* src = g + (size_t)(row0 + r) * ld + (k0 + c0);
  __bf16* dst = s + r * RS + c0;
#if HAS_ASYNC_LDS
  __builtin_amdgcn_global_load_async_to_lds_b128((gv4i_p)src,       (lv4i_p)dst,       0, 0);
  __builtin_amdgcn_global_load_async_to_lds_b128((gv4i_p)(src + 8), (lv4i_p)(dst + 8), 0, 0);
#else
  v8bf h0 = *(const v8bf*)(src);
  v8bf h1 = *(const v8bf*)(src + 8);
  *(v8bf*)(dst)     = h0;
  *(v8bf*)(dst + 8) = h1;
#endif
}

__device__ __forceinline__ void async_wait0() {
#if HAS_ASYNC_LDS
  asm volatile("s_wait_asynccnt 0x0" ::: "memory");
#endif
}

// ---------------------------------------------------------------------------
// Load a 16x32 bf16 WMMA A/B fragment from an LDS tile.
// ISA layout (16-bit, 16x32): lane L<16 holds row L, K in {0..7, 16..23};
// lane L+16 holds row L, K in {8..15, 24..31}.  (For B = W^T with W row-major,
// lane n streams row n of W.)
// ---------------------------------------------------------------------------
__device__ __forceinline__ v16bf frag_ld(const __bf16* s, int rowBase, int lane) {
  const int r  = rowBase + (lane & 15);
  const int hs = (lane >> 4) << 3;   // 0 or 8
  const __bf16* p = s + r * RS + hs;
  union { v16bf v; v8bf h[2]; } u;
  u.h[0] = *(const v8bf*)(p);        // ds_load_b128
  u.h[1] = *(const v8bf*)(p + 16);
  return u.v;
}

// ---------------------------------------------------------------------------
// Fused GEMM + epilogue (MODE is compile-time -> branch-free hot loop).
//  MODE 0: acc = x@W_ih^T ; v = acc+b_ih+b_hh ; xproj=v ; out[:,0,:]=relu(v)
//  MODE 1: acc = h@W_hh^T ; v = relu(xproj+acc) ; out[:,t,:]=v
// Both: hout = bf16(v).
// Block tile 128(M) x 128(N); 8 waves in 4(M) x 2(N); wave tile 32x64 = 2x4 WMMA.
// LDS double-buffered; async stage of k+1 overlaps WMMA on k.
// ---------------------------------------------------------------------------
template <int MODE>
__global__ void __launch_bounds__(256)
rnn_gemm(const __bf16* __restrict__ A, const __bf16* __restrict__ Bw,
         const float* __restrict__ bih, const float* __restrict__ bhh,
         float* __restrict__ xproj, float* __restrict__ out,
         __bf16* __restrict__ hout, int H, int T, int t) {
  __shared__ alignas(16) __bf16 sA[2][128 * RS];
  __shared__ alignas(16) __bf16 sB[2][128 * RS];

  const int tid  = threadIdx.x;
  const int lane = tid & 31;
  const int wave = tid >> 5;
  const int wm   = wave & 3;    // M sub-offset: wm*32
  const int wn   = wave >> 2;   // N sub-offset: wn*64
  const int mRow0 = blockIdx.x * 128;
  const int nRow0 = blockIdx.y * 128;

  v8f acc[2][4] = {};

  // prologue: fill buffer 0
  stage_tile(A,  H, mRow0, 0, sA[0], tid);
  stage_tile(Bw, H, nRow0, 0, sB[0], tid);
  async_wait0();
  __syncthreads();

  int cur = 0;
  for (int k0 = 0; k0 < H; k0 += 32) {
    const int nxt = cur ^ 1;
    if (k0 + 32 < H) {   // uniform branch; async copy overlaps compute below
      stage_tile(A,  H, mRow0, k0 + 32, sA[nxt], tid);
      stage_tile(Bw, H, nRow0, k0 + 32, sB[nxt], tid);
    }

    v16bf afr[2], bfr[4];
    afr[0] = frag_ld(sA[cur], wm * 32 + 0,  lane);
    afr[1] = frag_ld(sA[cur], wm * 32 + 16, lane);
    bfr[0] = frag_ld(sB[cur], wn * 64 + 0,  lane);
    bfr[1] = frag_ld(sB[cur], wn * 64 + 16, lane);
    bfr[2] = frag_ld(sB[cur], wn * 64 + 32, lane);
    bfr[3] = frag_ld(sB[cur], wn * 64 + 48, lane);

#pragma unroll
    for (int mi = 0; mi < 2; ++mi)
#pragma unroll
      for (int ni = 0; ni < 4; ++ni)
        acc[mi][ni] = __builtin_amdgcn_wmma_f32_16x16x32_bf16(
            false, afr[mi], false, bfr[ni], (short)0, acc[mi][ni], false, false);

    async_wait0();      // next buffer landed
    __syncthreads();    // all waves done reading cur / writing nxt
    cur = nxt;
  }

  // Epilogue.  C/D layout: VGPR r of lane l -> M = r + 8*(l>>4), N = l&15.
  const int ln = lane & 15;
  const int lh = lane >> 4;
#pragma unroll
  for (int mi = 0; mi < 2; ++mi) {
#pragma unroll
    for (int ni = 0; ni < 4; ++ni) {
      const int gN = nRow0 + wn * 64 + ni * 16 + ln;
      float bias = 0.f;
      if (MODE == 0) bias = bih[gN] + bhh[gN];
      v8f a = acc[mi][ni];
#pragma unroll
      for (int i = 0; i < 8; ++i) {
        const int gM = mRow0 + wm * 32 + mi * 16 + lh * 8 + i;
        const size_t idx = (size_t)gM * H + gN;
        float v;
        if (MODE == 0) {
          v = a[i] + bias;
          xproj[idx] = v;
          v = fmaxf(v, 0.f);
        } else {
          v = fmaxf(xproj[idx] + a[i], 0.f);
        }
        out[((size_t)gM * T + t) * H + gN] = v;
        hout[idx] = (__bf16)v;
      }
    }
  }
}

// ---------------------------------------------------------------------------
// kernel_launch: inputs = [input, W_ih, W_hh, b_ih, b_hh, n_instructions]
// ---------------------------------------------------------------------------
extern "C" void kernel_launch(void* const* d_in, const int* in_sizes, int n_in,
                              void* d_out, int out_size, void* d_ws, size_t ws_size,
                              hipStream_t stream) {
  const float* x   = (const float*)d_in[0];
  const float* Wih = (const float*)d_in[1];
  const float* Whh = (const float*)d_in[2];
  const float* bih = (const float*)d_in[3];
  const float* bhh = (const float*)d_in[4];

  const int H = in_sizes[3];                       // hidden size (b_ih length)
  const long long Bm = (long long)in_sizes[0] / H; // batch rows
  const int T = out_size / in_sizes[0];            // steps (avoid device scalar read)

  float* out = (float*)d_out;

  // workspace: xproj fp32 | hA bf16 | hB bf16 | W_hh bf16 | W_ih bf16 | x bf16
  char*   ws    = (char*)d_ws;
  float*  xproj = (float*)ws;
  __bf16* hA    = (__bf16*)(ws + (size_t)Bm * H * sizeof(float));
  __bf16* hB    = hA   + (size_t)Bm * H;
  __bf16* whhb  = hB   + (size_t)Bm * H;
  __bf16* wihb  = whhb + (size_t)H * H;
  __bf16* xb    = wihb + (size_t)H * H;

  const int nW = H * H;
  const long long nX = Bm * H;
  cvt_f32_bf16<<<(nW / 4 + 255) / 256, 256, 0, stream>>>(Whh, whhb, nW);
  cvt_f32_bf16<<<(nW / 4 + 255) / 256, 256, 0, stream>>>(Wih, wihb, nW);
  cvt_f32_bf16<<<(unsigned)((nX / 4 + 255) / 256), 256, 0, stream>>>(x, xb, (int)nX);

  dim3 grid((unsigned)(Bm / 128), (unsigned)(H / 128));

  // step 0: h1 = relu(x @ W_ih^T + b_ih + b_hh), stash xproj
  rnn_gemm<0><<<grid, 256, 0, stream>>>(xb, wihb, bih, bhh, xproj, out, hA, H, T, 0);

  __bf16* hin = hA;
  __bf16* hot = hB;
  for (int t = 1; t < T; ++t) {
    rnn_gemm<1><<<grid, 256, 0, stream>>>(hin, whhb, nullptr, nullptr,
                                          xproj, out, hot, H, T, t);
    __bf16* tmp = hin; hin = hot; hot = tmp;
  }
}
